// Attention_87608742904120
// MI455X (gfx1250) — matrix operand order repaired and checked by reading, not compiled, not run on hardware
//
#include <hip/hip_runtime.h>

// ---- sizes ----
#define SEQ 4096
#define DIM 1024
#define NH  16
#define HD  64

typedef _Float16 h8  __attribute__((ext_vector_type(8)));
typedef _Float16 h16 __attribute__((ext_vector_type(16)));
typedef __fp16   fp16x2 __attribute__((ext_vector_type(2)));   // return type of cvt_pkrtz
typedef float    f8  __attribute__((ext_vector_type(8)));

// ---- WMMA fragment loaders (16-bit, wave32; contiguous k order used alike for A and B) ----
// A (16xK=32, lane m = lane&15): K = 16*half + {0..15} contiguous
__device__ __forceinline__ h16 loadA(const _Float16* p, int half) {
    union { h16 v; h8 h[2]; } u;
    u.h[0] = *(const h8*)(p + 16 * half);
    u.h[1] = *(const h8*)(p + 16 * half + 8);
    return u.v;
}
// B (K=32 x 16, lane n = lane&15 holds column n, K = 16*half + {0..15} contiguous)
__device__ __forceinline__ h16 loadB(const _Float16* p, int half) {
    union { h16 v; h8 h[2]; } u;
    u.h[0] = *(const h8*)(p + 16 * half);
    u.h[1] = *(const h8*)(p + 16 * half + 8);
    return u.v;
}

#define WMMA(a, b, c) __builtin_amdgcn_wmma_f32_16x16x32_f16(false, (a), false, (b), (short)0, (c), false, false)

// ---- CDNA5 async global->LDS copy (16B per lane), ASYNCcnt-tracked ----
__device__ __forceinline__ void async_copy_b128(const _Float16* g, _Float16* lds) {
    unsigned loff = (unsigned)(unsigned long long)(const void*)lds;  // LDS aperture: addr[31:0]
    asm volatile("global_load_async_to_lds_b128 %0, %1, off"
                 :: "v"(loff), "v"(g) : "memory");
}
__device__ __forceinline__ void wait_async0() {
    asm volatile("s_wait_asynccnt 0x0" ::: "memory");
}

// ---- pack kernels ----
__global__ void pack_x_kernel(const float* __restrict__ x, _Float16* __restrict__ xh) {
    int i = blockIdx.x * 256 + threadIdx.x;          // SEQ*DIM threads
    xh[i] = (_Float16)x[i];
}
// W: [DIM][ncols] row-major fp32  ->  wT: [ncols][DIM] f16 (ncols*DIM threads, DIM==1024)
__global__ void pack_wT_kernel(const float* __restrict__ W, _Float16* __restrict__ wT, int ncols) {
    int i = blockIdx.x * 256 + threadIdx.x;
    int n = i >> 10;           // / DIM
    int k = i & (DIM - 1);
    wT[i] = (_Float16)W[k * ncols + n];
}

// ============================================================================
// GEMM core: block = 256 rows x 64 cols; wave = 32 rows x 64 cols (8 wmma/it
// against one shared B tile). B tile staged in LDS by async DMA, double-
// buffered; A fragments register double-buffered. B ds loads batched so one
// s_wait_dscnt covers 8 wmmas.
// ============================================================================
#define GEMM_BODY(AROW0, AROW1, WT, CTILE)                                      \
    __shared__ alignas(16) _Float16 sB[2][64][32];                              \
    const int tid = threadIdx.x;                                                \
    const int bn = tid >> 2, bk8 = (tid & 3) * 8;                               \
    f8 acc0[4] = {};                                                            \
    f8 acc1[4] = {};                                                            \
    async_copy_b128(WT + (size_t)(CTILE + bn) * DIM + bk8, &sB[0][bn][bk8]);    \
    h16 a0 = loadA(AROW0, half);                                                \
    h16 a1 = loadA(AROW1, half);                                                \
    int cur = 0;                                                                \
    for (int kb = 0; kb < DIM; kb += 32) {                                      \
        wait_async0();                                                          \
        __syncthreads();                                                        \
        if (kb + 32 < DIM)                                                      \
            async_copy_b128(WT + (size_t)(CTILE + bn) * DIM + kb + 32 + bk8,    \
                            &sB[cur ^ 1][bn][bk8]);                             \
        h16 a0n = a0, a1n = a1;                                                 \
        if (kb + 32 < DIM) {                                                    \
            a0n = loadA(AROW0 + kb + 32, half);                                 \
            a1n = loadA(AROW1 + kb + 32, half);                                 \
        }                                                                       \
        h16 b0 = loadB(&sB[cur][ln][0], half);                                  \
        h16 b1 = loadB(&sB[cur][16 + ln][0], half);                             \
        h16 b2 = loadB(&sB[cur][32 + ln][0], half);                             \
        h16 b3 = loadB(&sB[cur][48 + ln][0], half);                             \
        acc0[0] = WMMA(a0, b0, acc0[0]);  acc1[0] = WMMA(a1, b0, acc1[0]);      \
        acc0[1] = WMMA(a0, b1, acc0[1]);  acc1[1] = WMMA(a1, b1, acc1[1]);      \
        acc0[2] = WMMA(a0, b2, acc0[2]);  acc1[2] = WMMA(a1, b2, acc1[2]);      \
        acc0[3] = WMMA(a0, b3, acc0[3]);  acc1[3] = WMMA(a1, b3, acc1[3]);      \
        a0 = a0n; a1 = a1n;                                                     \
        cur ^= 1;                                                               \
    }

// ---- QKV GEMM: [SEQ,DIM] x [DIM,3*DIM] + b -> q (scaled), k, vT ----
__global__ __launch_bounds__(256) void qkv_gemm_kernel(
        const _Float16* __restrict__ xh, const _Float16* __restrict__ wT,
        const float* __restrict__ bias,
        _Float16* __restrict__ qh, _Float16* __restrict__ kh, _Float16* __restrict__ vT) {
    const int lane = threadIdx.x & 31, wid = threadIdx.x >> 5;
    const int half = lane >> 4, ln = lane & 15;
    const int rowbase = (blockIdx.x * 8 + wid) * 32;
    const int ctile = blockIdx.y * 64;
    const _Float16* arow0 = xh + (size_t)(rowbase + ln) * DIM;
    const _Float16* arow1 = xh + (size_t)(rowbase + 16 + ln) * DIM;
    GEMM_BODY(arow0, arow1, wT, ctile)
#pragma unroll
    for (int s = 0; s < 4; ++s) {
        const int cg = ctile + s * 16 + ln;          // global column in [0,3*DIM)
        const float bv = bias[cg];
        const int sel = cg >> 10;                    // 0:q 1:k 2:v
        const int cc = cg & (DIM - 1);
        const int head = cc >> 6, d = cc & (HD - 1);
#pragma unroll
        for (int g = 0; g < 2; ++g) {
            const f8& av = g ? acc1[s] : acc0[s];
#pragma unroll
            for (int v = 0; v < 8; ++v) {
                const int n = rowbase + g * 16 + v + 8 * half;
                const float val = av[v] + bv;
                if (sel == 0)      qh[(size_t)head * SEQ * HD + (size_t)n * HD + d] = (_Float16)(val * 0.125f);
                else if (sel == 1) kh[(size_t)head * SEQ * HD + (size_t)n * HD + d] = (_Float16)val;
                else               vT[(size_t)head * HD * SEQ + (size_t)d * SEQ + n] = (_Float16)val;
            }
        }
    }
}

// ---- output projection: [SEQ,DIM] x [DIM,DIM] + b -> fp32 out ----
__global__ __launch_bounds__(256) void proj_gemm_kernel(
        const _Float16* __restrict__ ah, const _Float16* __restrict__ wT,
        const float* __restrict__ bias, float* __restrict__ out) {
    const int lane = threadIdx.x & 31, wid = threadIdx.x >> 5;
    const int half = lane >> 4, ln = lane & 15;
    const int rowbase = (blockIdx.x * 8 + wid) * 32;
    const int ctile = blockIdx.y * 64;
    const _Float16* arow0 = ah + (size_t)(rowbase + ln) * DIM;
    const _Float16* arow1 = ah + (size_t)(rowbase + 16 + ln) * DIM;
    GEMM_BODY(arow0, arow1, wT, ctile)
#pragma unroll
    for (int s = 0; s < 4; ++s) {
        const int cg = ctile + s * 16 + ln;
        const float bv = bias[cg];
#pragma unroll
        for (int g = 0; g < 2; ++g) {
            const f8& av = g ? acc1[s] : acc0[s];
#pragma unroll
            for (int v = 0; v < 8; ++v) {
                const int n = rowbase + g * 16 + v + 8 * half;
                out[(size_t)n * DIM + cg] = av[v] + bv;
            }
        }
    }
}

// ============================================================================
// Flash attention: one wave = 16 query rows of one head. All 8 waves of a
// block share the head, so K/V tiles are staged once per block in LDS via
// async DMA (double-buffered). K-frag ds loads are batched; V-frag ds loads
// are hoisted above the softmax so their latency hides under v_exp work.
// ============================================================================
__global__ __launch_bounds__(256) void attn_kernel(
        const _Float16* __restrict__ qh, const _Float16* __restrict__ kh,
        const _Float16* __restrict__ vT, _Float16* __restrict__ oh) {
    __shared__ alignas(16) _Float16 sK[2][32][64];   // [buf][k-row][d]
    __shared__ alignas(16) _Float16 sV[2][64][32];   // [buf][d][k]
    const int tid = threadIdx.x;
    const int lane = tid & 31, wid = tid >> 5;
    const int half = lane >> 4, ln = lane & 15;
    const int wg = blockIdx.x * 8 + wid;             // 0 .. NH*SEQ/16 - 1
    const int head = wg >> 8;                        // same for all 8 waves of a block
    const int qbase = (wg & 255) * 16;

    const _Float16* kbp = kh + (size_t)head * SEQ * HD;
    const _Float16* vbp = vT + (size_t)head * HD * SEQ;

    // per-thread async-copy slices: K 32x64 halves, V 64x32 halves (16B each)
    const int krow = tid >> 3, kc8 = (tid & 7) * 8;
    const int vrow = tid >> 2, vk8 = (tid & 3) * 8;

    async_copy_b128(kbp + (size_t)krow * HD + kc8, &sK[0][krow][kc8]);
    async_copy_b128(vbp + (size_t)vrow * SEQ + vk8, &sV[0][vrow][vk8]);

    const _Float16* qp = qh + (size_t)head * SEQ * HD + (size_t)(qbase + ln) * HD;
    const h16 bq0 = loadB(qp, half);                 // d 0..31  (B: K=d)
    const h16 bq1 = loadB(qp + 32, half);            // d 32..63

    f8 ot[4] = {};                                   // O^T: d-chunk c rows x 16 queries
    float rm = -1e30f, rl = 0.0f;
    int cur = 0;

    for (int kb = 0; kb < SEQ; kb += 32) {
        wait_async0();
        __syncthreads();
        if (kb + 32 < SEQ) {
            async_copy_b128(kbp + (size_t)(kb + 32 + krow) * HD + kc8, &sK[cur ^ 1][krow][kc8]);
            async_copy_b128(vbp + (size_t)vrow * SEQ + kb + 32 + vk8, &sV[cur ^ 1][vrow][vk8]);
        }

        // ---- batched K fragment loads, then interleaved wmmas ----
        h16 ka00 = loadA(&sK[cur][ln][0], half);
        h16 ka01 = loadA(&sK[cur][ln][32], half);
        h16 ka10 = loadA(&sK[cur][16 + ln][0], half);
        h16 ka11 = loadA(&sK[cur][16 + ln][32], half);
        f8 st0 = {}, st1 = {};
        st0 = WMMA(ka00, bq0, st0);
        st1 = WMMA(ka10, bq0, st1);
        st0 = WMMA(ka01, bq1, st0);
        st1 = WMMA(ka11, bq1, st1);

        // ---- V fragments: issue ds loads now, consume after softmax ----
        h16 av0 = loadA(&sV[cur][ln][0], half);
        h16 av1 = loadA(&sV[cur][16 + ln][0], half);
        h16 av2 = loadA(&sV[cur][32 + ln][0], half);
        h16 av3 = loadA(&sV[cur][48 + ln][0], half);

        // ---- online softmax over k (16 in-lane values + one xor-16 reduce) ----
        float tmax = st0[0];
#pragma unroll
        for (int v = 1; v < 8; ++v) tmax = fmaxf(tmax, st0[v]);
#pragma unroll
        for (int v = 0; v < 8; ++v) tmax = fmaxf(tmax, st1[v]);
        tmax = fmaxf(tmax, __shfl_xor(tmax, 16, 32));
        const float mnew = fmaxf(rm, tmax);
        const float corr = __expf(rm - mnew);
        rm = mnew;

        float p0[8], p1[8], tsum = 0.0f;
#pragma unroll
        for (int v = 0; v < 8; ++v) { p0[v] = __expf(st0[v] - mnew); tsum += p0[v]; }
#pragma unroll
        for (int v = 0; v < 8; ++v) { p1[v] = __expf(st1[v] - mnew); tsum += p1[v]; }
        tsum += __shfl_xor(tsum, 16, 32);
        rl = rl * corr + tsum;
#pragma unroll
        for (int c = 0; c < 4; ++c)
#pragma unroll
            for (int v = 0; v < 8; ++v) ot[c][v] *= corr;

        // ---- build P^T as a WMMA B fragment (K=k tile, N=query) ----
        // lane(half) owns k = 16r + 8*half + v of frag r; B needs K = 16*half + {0..15}.
        unsigned own0[4], own1[4];
#pragma unroll
        for (int j = 0; j < 4; ++j) {
            union { fp16x2 h; unsigned u; } cv;
            cv.h = __builtin_amdgcn_cvt_pkrtz(p0[2 * j], p0[2 * j + 1]); own0[j] = cv.u;
            cv.h = __builtin_amdgcn_cvt_pkrtz(p1[2 * j], p1[2 * j + 1]); own1[j] = cv.u;
        }
        union { unsigned u[8]; h16 v; } pb;
#pragma unroll
        for (int j = 0; j < 4; ++j) {
            const unsigned pa0 = (unsigned)__shfl_xor((int)own0[j], 16, 32);
            const unsigned pa1 = (unsigned)__shfl_xor((int)own1[j], 16, 32);
            pb.u[j]     = half ? pa1 : own0[j];      // K = 16*half + {2j,2j+1}
            pb.u[4 + j] = half ? own1[j] : pa0;      // K = 16*half + 8 + {2j,2j+1}
        }
        const h16 pbf = pb.v;

        // ---- O^T += V^T (A, from LDS) x P^T (B) ----
        ot[0] = WMMA(av0, pbf, ot[0]);
        ot[1] = WMMA(av1, pbf, ot[1]);
        ot[2] = WMMA(av2, pbf, ot[2]);
        ot[3] = WMMA(av3, pbf, ot[3]);
        cur ^= 1;
    }

    const float inv = 1.0f / rl;
    _Float16* orow = oh + (size_t)(qbase + ln) * DIM + head * HD;
#pragma unroll
    for (int c = 0; c < 4; ++c)
#pragma unroll
        for (int v = 0; v < 8; ++v)
            orow[16 * c + 8 * half + v] = (_Float16)(ot[c][v] * inv);
}

extern "C" void kernel_launch(void* const* d_in, const int* in_sizes, int n_in,
                              void* d_out, int out_size, void* d_ws, size_t ws_size,
                              hipStream_t stream) {
    const float* x     = (const float*)d_in[0];
    const float* Wqkv  = (const float*)d_in[1];
    const float* bqkv  = (const float*)d_in[2];
    const float* Wproj = (const float*)d_in[3];
    const float* bproj = (const float*)d_in[4];
    float* out = (float*)d_out;

    _Float16* ws     = (_Float16*)d_ws;
    _Float16* xh     = ws;                                   // SEQ*DIM
    _Float16* wqkvT  = xh + (size_t)SEQ * DIM;               // 3*DIM*DIM
    _Float16* wprojT = wqkvT + (size_t)3 * DIM * DIM;        // DIM*DIM
    _Float16* qh     = wprojT + (size_t)DIM * DIM;           // SEQ*DIM
    _Float16* kh     = qh + (size_t)SEQ * DIM;               // SEQ*DIM
    _Float16* vT     = kh + (size_t)SEQ * DIM;               // SEQ*DIM (per-head transposed)
    _Float16* oh     = vT + (size_t)SEQ * DIM;               // SEQ*DIM

    pack_x_kernel<<<(SEQ * DIM) / 256, 256, 0, stream>>>(x, xh);
    pack_wT_kernel<<<(3 * DIM * DIM) / 256, 256, 0, stream>>>(Wqkv, wqkvT, 3 * DIM);
    pack_wT_kernel<<<(DIM * DIM) / 256, 256, 0, stream>>>(Wproj, wprojT, DIM);

    qkv_gemm_kernel<<<dim3(SEQ / 256, (3 * DIM) / 64), 256, 0, stream>>>(xh, wqkvT, bqkv, qh, kh, vT);
    attn_kernel<<<dim3((NH * SEQ / 16) / 8), 256, 0, stream>>>(qh, kh, vT, oh);
    proj_gemm_kernel<<<dim3(SEQ / 256, DIM / 64), 256, 0, stream>>>(oh, wprojT, bproj, out);
}
